// GL_GCNConv_9l_512h_w_44753559224351
// MI455X (gfx1250) — matrix operand order repaired and checked by reading, not compiled, not run on hardware
//
#include <hip/hip_runtime.h>
#include <hip/hip_bf16.h>
#include <math.h>

// ---------------------------------------------------------------------------
// Types for CDNA5 WMMA (wave32): D(16x16 f32) = A(16x32 bf16) x B(32x16 bf16) + C
// ---------------------------------------------------------------------------
typedef __attribute__((ext_vector_type(16))) __bf16 v16bf;
typedef __attribute__((ext_vector_type(8)))  float  v8f;
typedef __attribute__((ext_vector_type(4)))  float  f32x4;

union FragBF {
  v16bf v;
  f32x4 q[2];
};

__device__ __forceinline__ unsigned short f2bf(float f) {
  // round-to-nearest-even f32 -> bf16 (bit-level; avoids scalar __bf16 ops)
  unsigned int u = __float_as_uint(f);
  u += 0x7FFFu + ((u >> 16) & 1u);
  return (unsigned short)(u >> 16);
}

// ---------------------------------------------------------------------------
// GEMM: C[nrows x ncols] = A[nrows x K] (bf16, row major)
//                        x B (given as BT[ncolsPad x K] bf16 row-major = W^T)
// Wave tile: 32 (M) x 64 (N). Block: 256 threads = 8 waves covering 512 cols.
// ---------------------------------------------------------------------------
__global__ void __launch_bounds__(256)
k_gemm_wmma_bf16(const unsigned short* __restrict__ A,
                 const unsigned short* __restrict__ BT,
                 float* __restrict__ C,
                 int nrows, int K, int ncols, int ncolsPad)
{
  const int lane = threadIdx.x & 31;
  const int wave = threadIdx.x >> 5;
  const int m16  = lane & 15;
  const int hi   = lane >> 4;

  const int rowBase = blockIdx.x * 32;
  const int colBase = blockIdx.y * 512 + wave * 64;
  if (colBase >= ncolsPad) return;   // wave-uniform exit (EXEC stays all-ones)

  v8f acc[2][4];
  v8f zero = {};
#pragma unroll
  for (int mi = 0; mi < 2; ++mi)
#pragma unroll
    for (int ni = 0; ni < 4; ++ni)
      acc[mi][ni] = zero;

  // A-fragment row pointers (clamped so every lane stays in-bounds; results
  // from clamped rows are computed but never stored).
  int r0 = rowBase + m16;       if (r0 >= nrows) r0 = nrows - 1;
  int r1 = rowBase + 16 + m16;  if (r1 >= nrows) r1 = nrows - 1;
  const unsigned short* a0p = A + (size_t)r0 * K;
  const unsigned short* a1p = A + (size_t)r1 * K;

  const unsigned short* bp[4];
#pragma unroll
  for (int ni = 0; ni < 4; ++ni) {
    int c = colBase + ni * 16 + m16;
    if (c >= ncolsPad) c = ncolsPad - 1;
    bp[ni] = BT + (size_t)c * K;
  }

  for (int k0 = 0; k0 < K; k0 += 32) {
    FragBF a[2], b[4];
    // A layout (16-bit 16x32): lane(m,hi): elems 0..7 -> K = 8*hi+e,
    //                          elems 8..15 -> K = 8*hi+16+(e-8)
    a[0].q[0] = *(const f32x4*)(a0p + k0 + 8 * hi);
    a[0].q[1] = *(const f32x4*)(a0p + k0 + 8 * hi + 16);
    a[1].q[0] = *(const f32x4*)(a1p + k0 + 8 * hi);
    a[1].q[1] = *(const f32x4*)(a1p + k0 + 8 * hi + 16);
    // B layout (32x16): lane(col=m16,hi): elems 0..7 -> K = 8*hi+e,
    //                   elems 8..15 -> K = 8*hi+16+(e-8)
#pragma unroll
    for (int ni = 0; ni < 4; ++ni) {
      b[ni].q[0] = *(const f32x4*)(bp[ni] + k0 + 8 * hi);
      b[ni].q[1] = *(const f32x4*)(bp[ni] + k0 + 8 * hi + 16);
    }
#pragma unroll
    for (int mi = 0; mi < 2; ++mi)
#pragma unroll
      for (int ni = 0; ni < 4; ++ni)
        acc[mi][ni] = __builtin_amdgcn_wmma_f32_16x16x32_bf16(
            false, a[mi].v, false, b[ni].v,
            (short)0, acc[mi][ni], false, false);
  }

  // D layout (f32 16x16): lane(n=m16,hi): VGPR r -> row M = 8*hi + r, col n
#pragma unroll
  for (int mi = 0; mi < 2; ++mi) {
    int row0 = rowBase + mi * 16 + hi * 8;
#pragma unroll
    for (int ni = 0; ni < 4; ++ni) {
      int col = colBase + ni * 16 + m16;
      if (col >= ncols) continue;
#pragma unroll
      for (int r = 0; r < 8; ++r) {
        int row = row0 + r;
        if (row < nrows)
          C[(size_t)row * ncols + col] = acc[mi][ni][r];
      }
    }
  }
}

// ---------------------------------------------------------------------------
// Elementwise / graph kernels
// ---------------------------------------------------------------------------
__global__ void __launch_bounds__(256)
k_fill1(float* __restrict__ p, int n) {
  int i = blockIdx.x * blockDim.x + threadIdx.x;
  if (i < n) p[i] = 1.0f;   // self-loop weight folded into degree init
}

__global__ void __launch_bounds__(256)
k_deg_acc(const long long* __restrict__ colI, const float* __restrict__ w,
          float* __restrict__ deg, int E) {
  int i = blockIdx.x * blockDim.x + threadIdx.x;
  if (i < E) atomicAdd(&deg[(int)colI[i]], w[i]);
}

__global__ void __launch_bounds__(256)
k_dinv(const float* __restrict__ deg, float* __restrict__ dinv,
       float* __restrict__ nself, int n) {
  int i = blockIdx.x * blockDim.x + threadIdx.x;
  if (i >= n) return;
  float d = deg[i];
  float r = d > 0.0f ? rsqrtf(d) : 0.0f;
  dinv[i]  = r;
  nself[i] = r * r;   // self-loop norm: dinv[i] * 1 * dinv[i]
}

__global__ void __launch_bounds__(256)
k_norm(const long long* __restrict__ rowI, const long long* __restrict__ colI,
       const float* __restrict__ w, const float* __restrict__ dinv,
       float* __restrict__ norm, int E) {
  int i = blockIdx.x * blockDim.x + threadIdx.x;
  if (i < E) norm[i] = dinv[(int)rowI[i]] * w[i] * dinv[(int)colI[i]];
}

__global__ void __launch_bounds__(256)
k_f32_to_bf16(const float* __restrict__ src, unsigned short* __restrict__ dst,
              int total) {
  int i = blockIdx.x * blockDim.x + threadIdx.x;
  if (i < total) dst[i] = f2bf(src[i]);
}

// WT[n*dIn + k] = bf16(W[k*dOut + n]); zero-pad n in [dOut, dOutPad)
__global__ void __launch_bounds__(256)
k_wt_pack(const float* __restrict__ W, unsigned short* __restrict__ WT,
          int dOut, int total, int dInShift, int dInMask) {
  int idx = blockIdx.x * blockDim.x + threadIdx.x;
  if (idx >= total) return;
  int n = idx >> dInShift;
  int k = idx & dInMask;
  float v = (n < dOut) ? W[(size_t)k * dOut + n] : 0.0f;
  WT[idx] = f2bf(v);
}

// agg[n,f] = bias[f] + normSelf[n] * m[n,f]   (self-loop handled atomic-free)
__global__ void __launch_bounds__(256)
k_agg_init(const float* __restrict__ m, const float* __restrict__ nself,
           const float* __restrict__ bias, float* __restrict__ agg,
           int total, int sh, int mask) {
  int idx = blockIdx.x * blockDim.x + threadIdx.x;
  if (idx >= total) return;
  int n = idx >> sh;
  int f = idx & mask;
  agg[idx] = bias[f] + nself[n] * m[idx];
}

// agg[col[e], f] += norm[e] * m[row[e], f]; one thread per (edge, feature)
__global__ void __launch_bounds__(256)
k_scatter(const long long* __restrict__ rowI, const long long* __restrict__ colI,
          const float* __restrict__ norm, const float* __restrict__ m,
          float* __restrict__ agg, int total, int sh, int mask) {
  int idx = blockIdx.x * blockDim.x + threadIdx.x;
  if (idx >= total) return;
  int e = idx >> sh;
  int f = idx & mask;
  int src = (int)rowI[e];
  int dst = (int)colI[e];
  float v = norm[e] * m[((size_t)src << sh) + f];
  atomicAdd(agg + ((size_t)dst << sh) + f, v);
}

__global__ void __launch_bounds__(256)
k_elu_bf16(const float* __restrict__ agg, unsigned short* __restrict__ hB,
           int total) {
  int i = blockIdx.x * blockDim.x + threadIdx.x;
  if (i >= total) return;
  float v = agg[i];
  v = v > 0.0f ? v : expm1f(v);
  hB[i] = f2bf(v);
}

// ---------------------------------------------------------------------------
// Orchestration
// ---------------------------------------------------------------------------
static inline int ilog2i(int v) { int s = 0; while ((1 << s) < v) ++s; return s; }

extern "C" void kernel_launch(void* const* d_in, const int* in_sizes, int n_in,
                              void* d_out, int out_size, void* d_ws, size_t ws_size,
                              hipStream_t stream) {
  (void)n_in; (void)out_size; (void)ws_size;

  const int F_IN = 128;
  const int N = in_sizes[0] / F_IN;       // 50000
  const int E = in_sizes[2];              // 800000

  const float*     x  = (const float*)d_in[0];
  const long long* ei = (const long long*)d_in[1];   // int64 [2, E]
  const float*     ew = (const float*)d_in[2];
  const float* Ws[9];
  const float* bs[9];
  for (int i = 0; i < 9; ++i) {
    Ws[i] = (const float*)d_in[3 + i];
    bs[i] = (const float*)d_in[12 + i];
  }
  const long long* rowI = ei;        // message source j
  const long long* colI = ei + E;    // aggregation target i

  // ---- scratch carve (256B aligned) ----
  size_t off = 0;
  char* base = (char*)d_ws;
  auto take = [&](size_t bytes) -> char* {
    char* p = base + off;
    off += (bytes + 255) & ~(size_t)255;
    return p;
  };
  unsigned short* hB    = (unsigned short*)take((size_t)N * 512 * 2); // bf16 acts
  float*          m     = (float*)take((size_t)N * 512 * 4);          // h @ W
  float*          agg   = (float*)take((size_t)N * 512 * 4);
  float*          deg   = (float*)take((size_t)N * 4);
  float*          dinv  = (float*)take((size_t)N * 4);
  float*          nself = (float*)take((size_t)N * 4);
  float*          norm  = (float*)take((size_t)E * 4);
  unsigned short* WT    = (unsigned short*)take((size_t)512 * 512 * 2);

  const int TPB = 256;
  auto nb = [&](long long total) { return (unsigned int)((total + TPB - 1) / TPB); };

  // ---- normalization (once) ----
  k_fill1<<<nb(N), TPB, 0, stream>>>(deg, N);
  k_deg_acc<<<nb(E), TPB, 0, stream>>>(colI, ew, deg, E);
  k_dinv<<<nb(N), TPB, 0, stream>>>(deg, dinv, nself, N);
  k_norm<<<nb(E), TPB, 0, stream>>>(rowI, colI, ew, dinv, norm, E);

  // ---- input activations -> bf16 ----
  k_f32_to_bf16<<<nb((long long)N * F_IN), TPB, 0, stream>>>(x, hB, N * F_IN);

  const int dims[10] = {128, 512, 512, 512, 512, 512, 512, 512, 512, 8};

  for (int l = 0; l < 9; ++l) {
    const int dIn     = dims[l];
    const int dOut    = dims[l + 1];
    const int dOutPad = (dOut < 16) ? 16 : dOut;
    const int shIn    = ilog2i(dIn);
    const int shOut   = ilog2i(dOut);

    // pack W^T (bf16, padded)
    k_wt_pack<<<nb((long long)dOutPad * dIn), TPB, 0, stream>>>(
        Ws[l], WT, dOut, dOutPad * dIn, shIn, dIn - 1);

    // m = h @ W  (WMMA bf16 -> f32)
    dim3 gg((N + 31) / 32, (dOutPad + 511) / 512);
    k_gemm_wmma_bf16<<<gg, 256, 0, stream>>>(hB, WT, m, N, dIn, dOut, dOutPad);

    // aggregate: bias + self-loop, then edge scatter-add
    float* aggPtr = (l == 8) ? (float*)d_out : agg;
    k_agg_init<<<nb((long long)N * dOut), TPB, 0, stream>>>(
        m, nself, bs[l], aggPtr, N * dOut, shOut, dOut - 1);
    k_scatter<<<nb((long long)E * dOut), TPB, 0, stream>>>(
        rowI, colI, norm, m, aggPtr, E * dOut, shOut, dOut - 1);

    // activation + bf16 for next layer
    if (l < 8)
      k_elu_bf16<<<nb((long long)N * dOut), TPB, 0, stream>>>(aggPtr, hB, N * dOut);
  }
}
